// RAN_420906795334
// MI455X (gfx1250) — matrix operand order repaired and checked by reading, not compiled, not run on hardware
//
#include <hip/hip_runtime.h>
#include <hip/hip_bf16.h>
#include <math.h>

// RAN forward: T=128, B=32, H=1024, V=32000
#define T_STEPS 128
#define BATCH   32
#define HID     1024
#define VOCAB   32000
#define ROWS    (T_STEPS * BATCH)   // 4096

typedef __attribute__((ext_vector_type(16))) __bf16 v16bf;
typedef __attribute__((ext_vector_type(8)))  __bf16 v8bf;
typedef __attribute__((ext_vector_type(8)))  float  v8f;

// ---------------------------------------------------------------------------
// f32 -> bf16 cast (grid-stride)
// ---------------------------------------------------------------------------
__global__ __launch_bounds__(256)
void cast_f32_bf16(const float* __restrict__ src, __bf16* __restrict__ dst, long n)
{
    long i = (long)blockIdx.x * blockDim.x + threadIdx.x;
    long stride = (long)gridDim.x * blockDim.x;
    for (; i < n; i += stride) dst[i] = (__bf16)src[i];
}

// async DMA: 16 bytes global -> LDS, tracked by ASYNCcnt
__device__ __forceinline__ void async_b128(unsigned lds_addr, const __bf16* g)
{
    unsigned long long ga = (unsigned long long)(size_t)g;
    asm volatile("global_load_async_to_lds_b128 %0, %1, off"
                 :: "v"(lds_addr), "v"(ga) : "memory");
}
#define WAIT_ASYNC1() asm volatile("s_wait_asynccnt 0x1" ::: "memory")
#define WAIT_ASYNC0() asm volatile("s_wait_asynccnt 0x0" ::: "memory")

// ---------------------------------------------------------------------------
// WMMA GEMM: C[m,n] = sum_k A[m,k] * W[n,k] (+ bias[n])
// A: M x K bf16 row-major (lda = K). W: N x K bf16 row-major (ldw = K).
// Block = 256 thr = 8 waves. Block tile 128(M) x 64(N); wave tile 16 x 64.
// B tiles (32K x 64N) staged via ASYNC global->LDS DMA, statically double
// buffered, software pipelined with 2 copies in flight; steady-state loop
// has no guards or selects. Requires M%128==0, N%64==0, K%64==0, K>=128.
// ---------------------------------------------------------------------------
__global__ __launch_bounds__(256)
void wmma_gemm_bias(const __bf16* __restrict__ A, const __bf16* __restrict__ W,
                    const float* __restrict__ bias, float* __restrict__ C,
                    int K, int ldc)
{
    __shared__ __bf16 ldsB[2][64 * 32];       // [buf][n_local][k_local]

    const int tid  = threadIdx.x;
    const int lane = tid & 31;
    const int wave = tid >> 5;
    const int n0   = blockIdx.x * 64;
    const int m0   = blockIdx.y * 128 + wave * 16;
    const int half = lane >> 4;               // 0/1 lane-group
    const int l15  = lane & 15;
    const long arow  = (long)(m0 + l15) * K;  // A row for this lane
    const int  akoff = half * 16;             // A K-half offset (contiguous k = 16*half + s)
    const int  bkoff = half * 16;              // B K-half offset (contiguous k = 16*half + s)

    // staging assignment: each thread DMAs one 16B chunk of the B tile
    const int  sn   = tid >> 2;               // 0..63 (n within tile)
    const int  sk   = (tid & 3) * 8;          // 0,8,16,24 (k within tile)
    const __bf16* wsrc = W + (long)(n0 + sn) * K + sk;

    // LDS byte addresses for this thread's chunk (low 32 bits of generic ptr)
    const unsigned lds0 = (unsigned)(size_t)(&ldsB[0][sn * 32 + sk]);
    const unsigned lds1 = (unsigned)(size_t)(&ldsB[1][sn * 32 + sk]);

    v8f acc[4] = {};

    auto compute = [&](const __bf16* bbase, int kk) {
        __builtin_prefetch(A + arow + kk + 256, 0, 1);
        v8bf alo = *(const v8bf*)(A + arow + kk + akoff);
        v8bf ahi = *(const v8bf*)(A + arow + kk + 8 + akoff);
        v16bf a = __builtin_shufflevector(alo, ahi,
                    0,1,2,3,4,5,6,7,8,9,10,11,12,13,14,15);
#pragma unroll
        for (int j = 0; j < 4; ++j) {
            v16bf b = *(const v16bf*)(bbase + (j * 16 + l15) * 32 + bkoff);
            acc[j] = __builtin_amdgcn_wmma_f32_16x16x32_bf16(
                         false, a, false, b, (short)0, acc[j], false, false);
        }
    };

    // prologue: 2 async tile copies in flight
    async_b128(lds0, wsrc);
    async_b128(lds1, wsrc + 32);

    int kk = 0;
    for (; kk < K - 64; kk += 64) {
        WAIT_ASYNC1();                 // tile kk landed (kk+32 still flying)
        __syncthreads();
        compute(&ldsB[0][0], kk);
        __syncthreads();               // all waves done reading buf0
        async_b128(lds0, wsrc + kk + 64);

        WAIT_ASYNC1();                 // tile kk+32 landed (kk+64 flying)
        __syncthreads();
        compute(&ldsB[1][0], kk + 32);
        __syncthreads();               // all waves done reading buf1
        async_b128(lds1, wsrc + kk + 96);
    }
    // epilogue: consume final two tiles (kk == K-64 here)
    WAIT_ASYNC1();
    __syncthreads();
    compute(&ldsB[0][0], kk);
    __syncthreads();
    WAIT_ASYNC0();
    __syncthreads();
    compute(&ldsB[1][0], kk + 32);

#pragma unroll
    for (int j = 0; j < 4; ++j) {
        const int col = n0 + j * 16 + l15;
        const float bv = bias ? bias[col] : 0.0f;
#pragma unroll
        for (int v = 0; v < 8; ++v) {
            const int row = m0 + half * 8 + v;
            C[(long)row * ldc + col] = acc[j][v] + bv;
        }
    }
}

// ---------------------------------------------------------------------------
// One recurrence step:
//   i = sigmoid(c_prev @ w_ic^T + b_i + ix_t)
//   f = sigmoid(c_prev @ w_fc^T + b_f + fx_t)
//   c = i * x_t + f * c_prev
// A (bf16 c_prev): 32 x 1024. Block = 64 thr = 2 waves (M strips), 64 N cols.
// Grid.x = HID/64 = 16.
// ---------------------------------------------------------------------------
__global__ __launch_bounds__(64)
void ran_step(const __bf16* __restrict__ Aprev,
              const __bf16* __restrict__ Wic, const __bf16* __restrict__ Wfc,
              const float* __restrict__ bi,  const float* __restrict__ bfv,
              const float* __restrict__ ixt, const float* __restrict__ fxt,
              const float* __restrict__ xt,  const float* __restrict__ cprev,
              float* __restrict__ cnew, __bf16* __restrict__ cnew_bf)
{
    const int lane = threadIdx.x & 31;
    const int wave = threadIdx.x >> 5;
    const int n0   = blockIdx.x * 64;
    const int m0   = wave * 16;
    const int half = lane >> 4;
    const int l15  = lane & 15;
    const long arow  = (long)(m0 + l15) * HID;
    const int  akoff = half * 16;
    const int  bkoff = half * 16;

    v8f ai[4] = {};
    v8f af[4] = {};

    for (int kk = 0; kk < HID; kk += 32) {
        v8bf alo = *(const v8bf*)(Aprev + arow + kk + akoff);
        v8bf ahi = *(const v8bf*)(Aprev + arow + kk + 8 + akoff);
        v16bf a = __builtin_shufflevector(alo, ahi,
                    0,1,2,3,4,5,6,7,8,9,10,11,12,13,14,15);
#pragma unroll
        for (int j = 0; j < 4; ++j) {
            const long wroff = (long)(n0 + j * 16 + l15) * HID + kk + bkoff;
            v16bf bi16 = *(const v16bf*)(Wic + wroff);
            ai[j] = __builtin_amdgcn_wmma_f32_16x16x32_bf16(
                        false, a, false, bi16, (short)0, ai[j], false, false);
            v16bf bf16 = *(const v16bf*)(Wfc + wroff);
            af[j] = __builtin_amdgcn_wmma_f32_16x16x32_bf16(
                        false, a, false, bf16, (short)0, af[j], false, false);
        }
    }

#pragma unroll
    for (int j = 0; j < 4; ++j) {
        const int col = n0 + j * 16 + l15;
        const float bic = bi[col];
        const float bfc = bfv[col];
#pragma unroll
        for (int v = 0; v < 8; ++v) {
            const int row = m0 + half * 8 + v;
            const long idx = (long)row * HID + col;
            float pi = ai[j][v] + bic + ixt[idx];
            float pf = af[j][v] + bfc + fxt[idx];
            float ig = 1.0f / (1.0f + expf(-pi));
            float fg = 1.0f / (1.0f + expf(-pf));
            float cn = ig * xt[idx] + fg * cprev[idx];
            cnew[idx]    = cn;
            cnew_bf[idx] = (__bf16)cn;
        }
    }
}

// ---------------------------------------------------------------------------
// Per-row online max / log-sum-exp over V. One block (256 thr) per row.
// stats[row*2] = max, stats[row*2+1] = log(sum exp(x - max))
// ---------------------------------------------------------------------------
__global__ __launch_bounds__(256)
void row_logsumexp(const float* __restrict__ logits, float* __restrict__ stats)
{
    const int row = blockIdx.x;
    const int tid = threadIdx.x;
    const float* p = logits + (long)row * VOCAB;

    float m = -INFINITY, s = 0.0f;
    for (int i = tid; i < VOCAB; i += 256) {
        float x = p[i];
        if (x > m) { s = s * expf(m - x) + 1.0f; m = x; }
        else       { s += expf(x - m); }
    }
    __shared__ float sm[256], ss[256];
    sm[tid] = m; ss[tid] = s;
    __syncthreads();
    for (int off = 128; off > 0; off >>= 1) {
        if (tid < off) {
            float m2 = sm[tid + off], s2 = ss[tid + off];
            float mm = fmaxf(sm[tid], m2);
            ss[tid] = ss[tid] * expf(sm[tid] - mm) + s2 * expf(m2 - mm);
            sm[tid] = mm;
        }
        __syncthreads();
    }
    if (tid == 0) {
        stats[row * 2 + 0] = sm[0];
        stats[row * 2 + 1] = logf(ss[0]);
    }
}

// ---------------------------------------------------------------------------
// In-place: out = x - max - logsum. Grid: (chunks over V, rows).
// ---------------------------------------------------------------------------
__global__ __launch_bounds__(256)
void lsm_apply(float* __restrict__ logits, const float* __restrict__ stats)
{
    const int row = blockIdx.y;
    const float sub = stats[row * 2 + 0] + stats[row * 2 + 1];
    float* p = logits + (long)row * VOCAB;
    const int stride = gridDim.x * 256;
    for (int i = blockIdx.x * 256 + threadIdx.x; i < VOCAB; i += stride)
        p[i] -= sub;
}

// ---------------------------------------------------------------------------
extern "C" void kernel_launch(void* const* d_in, const int* in_sizes, int n_in,
                              void* d_out, int out_size, void* d_ws, size_t ws_size,
                              hipStream_t stream)
{
    (void)in_sizes; (void)n_in; (void)out_size; (void)ws_size;

    const float* x_in   = (const float*)d_in[0];   // [T,B,H]
    const float* hidden = (const float*)d_in[1];   // [B,H]
    const float* w_ic   = (const float*)d_in[2];
    const float* w_ix   = (const float*)d_in[3];
    const float* w_fc   = (const float*)d_in[4];
    const float* w_fx   = (const float*)d_in[5];
    const float* b_i    = (const float*)d_in[6];
    const float* b_f    = (const float*)d_in[7];
    const float* W_out  = (const float*)d_in[8];
    const float* b_out  = (const float*)d_in[9];
    float* out = (float*)d_out;                    // [T*B*V] log_probs + [B*H] hidden

    // workspace carve-up (all regions 256B aligned)
    char* w = (char*)d_ws;
    size_t off = 0;
    auto carve = [&](size_t bytes) {
        char* p = w + off;
        off += (bytes + 255) & ~(size_t)255;
        return p;
    };
    __bf16* inbf   = (__bf16*)carve((size_t)ROWS * HID * 2);     // input bf16
    __bf16* wicbf  = (__bf16*)carve((size_t)HID * HID * 2);
    __bf16* wfcbf  = (__bf16*)carve((size_t)HID * HID * 2);
    __bf16* wixbf  = (__bf16*)carve((size_t)HID * HID * 2);
    __bf16* wfxbf  = (__bf16*)carve((size_t)HID * HID * 2);
    __bf16* woutbf = (__bf16*)carve((size_t)VOCAB * HID * 2);
    float*  ixbuf  = (float*) carve((size_t)ROWS * HID * 4);
    float*  fxbuf  = (float*) carve((size_t)ROWS * HID * 4);
    __bf16* outsbf = (__bf16*)carve((size_t)ROWS * HID * 2);     // c_t history (bf16)
    __bf16* h0bf   = (__bf16*)carve((size_t)BATCH * HID * 2);
    float*  cf0    = (float*) carve((size_t)BATCH * HID * 4);
    float*  cf1    = (float*) carve((size_t)BATCH * HID * 4);
    float*  stats  = (float*) carve((size_t)ROWS * 2 * 4);

    // 1) casts to bf16
    cast_f32_bf16<<<2048, 256, 0, stream>>>(x_in,  inbf,   (long)ROWS * HID);
    cast_f32_bf16<<<512,  256, 0, stream>>>(w_ic,  wicbf,  (long)HID * HID);
    cast_f32_bf16<<<512,  256, 0, stream>>>(w_fc,  wfcbf,  (long)HID * HID);
    cast_f32_bf16<<<512,  256, 0, stream>>>(w_ix,  wixbf,  (long)HID * HID);
    cast_f32_bf16<<<512,  256, 0, stream>>>(w_fx,  wfxbf,  (long)HID * HID);
    cast_f32_bf16<<<4096, 256, 0, stream>>>(W_out, woutbf, (long)VOCAB * HID);
    cast_f32_bf16<<<64,   256, 0, stream>>>(hidden, h0bf,  (long)BATCH * HID);
    hipMemcpyAsync(cf0, hidden, (size_t)BATCH * HID * 4,
                   hipMemcpyDeviceToDevice, stream);

    // 2) input projections: ix = X @ w_ix^T, fx = X @ w_fx^T
    dim3 g1(HID / 64, ROWS / 128);
    wmma_gemm_bias<<<g1, 256, 0, stream>>>(inbf, wixbf, nullptr, ixbuf, HID, HID);
    wmma_gemm_bias<<<g1, 256, 0, stream>>>(inbf, wfxbf, nullptr, fxbuf, HID, HID);

    // 3) sequential recurrence (stream-ordered, 1 launch per step)
    float* cbuf[2] = { cf0, cf1 };
    for (int t = 0; t < T_STEPS; ++t) {
        const __bf16* Aprev = (t == 0) ? h0bf
                                       : outsbf + (size_t)(t - 1) * BATCH * HID;
        ran_step<<<HID / 64, 64, 0, stream>>>(
            Aprev, wicbf, wfcbf, b_i, b_f,
            ixbuf + (size_t)t * BATCH * HID,
            fxbuf + (size_t)t * BATCH * HID,
            x_in  + (size_t)t * BATCH * HID,
            cbuf[t & 1], cbuf[(t + 1) & 1],
            outsbf + (size_t)t * BATCH * HID);
    }
    // T even -> final state in cf0
    hipMemcpyAsync(out + (size_t)ROWS * VOCAB, cf0, (size_t)BATCH * HID * 4,
                   hipMemcpyDeviceToDevice, stream);

    // 4) logits = C @ W_out^T + b_out  (written into d_out)
    dim3 g2(VOCAB / 64, ROWS / 128);
    wmma_gemm_bias<<<g2, 256, 0, stream>>>(outsbf, woutbf, b_out, out, HID, VOCAB);

    // 5) log-softmax over V, in place on d_out
    row_logsumexp<<<ROWS, 256, 0, stream>>>(out, stats);
    dim3 g3(50, ROWS);
    lsm_apply<<<g3, 256, 0, stream>>>(out, stats);
}
